// MoE_12489764896830
// MI455X (gfx1250) — compile-verified, matrix operand order checked by reading, not run on hardware
//
#include <hip/hip_runtime.h>
#include <hip/hip_bf16.h>
#include <math.h>

// Problem constants (reference): D, F, E, k; T = 2*2048
constexpr int kD = 1024;
constexpr int kF = 2048;
constexpr int kE = 8;
constexpr int kT = 4096;
#define RMS_EPS 1.1920929e-07f

typedef __attribute__((ext_vector_type(16))) __bf16 v16bf;
typedef __attribute__((ext_vector_type(8)))  float  v8f;

union Frag {
    v16bf v;
    unsigned int u[8];
    __bf16 h[16];
};

// 16-bit A/B fragment K mapping (ISA 7.12.2): dword j of lane-half `hi`
// covers K = (j<4 ? 2j : 2j+8) + 8*hi   (two consecutive K per dword)
__device__ __forceinline__ int kmap_dword(int j, int hi) {
    return ((j < 4) ? 2 * j : 2 * j + 8) + hi * 8;
}
// half index j2 (0..15) -> K offset
__device__ __forceinline__ int kmap_half(int j2, int hi) {
    return ((j2 < 8) ? j2 : j2 + 8) + hi * 8;
}

__device__ __forceinline__ unsigned short bf16_bits(float f) {
    __bf16 b = (__bf16)f;
    return __builtin_bit_cast(unsigned short, b);
}

// ---------------------------------------------------------------------------
// Kernel 1: fused RMSNorm (fp32 -> bf16 xh) + gate scores + top-2 softmax.
// One wave32 per token; 8 waves (256 threads) per block.
// ---------------------------------------------------------------------------
__global__ __launch_bounds__(256) void gate_norm_kernel(
    const float* __restrict__ x, const float* __restrict__ Wg,
    __bf16* __restrict__ xh, float* __restrict__ pairGate,
    int* __restrict__ pairExpert)
{
    int wave = threadIdx.x >> 5;
    int lane = threadIdx.x & 31;
    int t = blockIdx.x * 8 + wave;
    if (t >= kT) return;

    const float* xr = x + (size_t)t * kD;
    float xs[32];
    float ss = 0.f;
    float sc[kE];
    #pragma unroll
    for (int e = 0; e < kE; e++) sc[e] = 0.f;

    #pragma unroll
    for (int i = 0; i < 32; i++) {
        int d = lane + i * 32;
        float v = xr[d];
        xs[i] = v;
        ss += v * v;
        #pragma unroll
        for (int e = 0; e < kE; e++) sc[e] += v * Wg[d * kE + e];
    }
    #pragma unroll
    for (int off = 16; off > 0; off >>= 1) {
        ss += __shfl_xor(ss, off, 32);
        #pragma unroll
        for (int e = 0; e < kE; e++) sc[e] += __shfl_xor(sc[e], off, 32);
    }

    float inv = rsqrtf(ss / (float)kD + RMS_EPS);
    #pragma unroll
    for (int i = 0; i < 32; i++) {
        xh[(size_t)t * kD + lane + i * 32] = (__bf16)(xs[i] * inv);
    }

    if (lane == 0) {
        int i0 = 0;
        for (int e = 1; e < kE; e++) if (sc[e] > sc[i0]) i0 = e;
        int i1 = (i0 == 0) ? 1 : 0;
        for (int e = 0; e < kE; e++) if (e != i0 && sc[e] > sc[i1]) i1 = e;
        float m  = sc[i0];
        float e0 = __expf(sc[i0] - m), e1 = __expf(sc[i1] - m);
        float z  = __builtin_amdgcn_rcpf(e0 + e1);
        pairExpert[t * 2 + 0] = i0;
        pairExpert[t * 2 + 1] = i1;
        pairGate[t * 2 + 0]   = e0 * z;
        pairGate[t * 2 + 1]   = e1 * z;
    }
}

// ---------------------------------------------------------------------------
// Kernel 2: scatter token-pairs into per-expert lists.
// ---------------------------------------------------------------------------
__global__ __launch_bounds__(256) void build_lists_kernel(
    const int* __restrict__ pairExpert, int* __restrict__ cnt,
    int* __restrict__ lists)
{
    int t = blockIdx.x * blockDim.x + threadIdx.x;
    if (t >= kT) return;
    #pragma unroll
    for (int s = 0; s < 2; s++) {
        int e   = pairExpert[t * 2 + s];
        int pos = atomicAdd(&cnt[e], 1);
        lists[e * kT + pos] = t * 2 + s;
    }
}

// ---------------------------------------------------------------------------
// Kernel 2b: pre-convert a weight matrix [E x K x N] fp32 -> bf16 in
// fragment-major ("swizzled") layout: per (e, tileN, tileK) a 32x16 tile
// stored as 1KB where lane l owns bytes [l*32, l*32+32) == its whole WMMA
// B fragment (two contiguous b128 loads in the GEMM). norm_w folded here.
// One thread per output dword (= one K-pair of one lane).
// ---------------------------------------------------------------------------
__global__ __launch_bounds__(256) void convert_weights_kernel(
    const float* __restrict__ src, const float* __restrict__ norm_w,
    unsigned int* __restrict__ dst, int K, int N)
{
    size_t idx = (size_t)blockIdx.x * 256 + threadIdx.x;
    size_t perE = (size_t)K * N / 2;            // dwords per expert
    if (idx >= perE * kE) return;

    int e = (int)(idx / perE);
    int d = (int)(idx % perE);
    int TK = K / 32;
    int tile = d >> 8;                          // 256 dwords per 32x16 tile
    int dw   = d & 255;
    int lane = dw >> 3;
    int j2   = (dw & 7) * 2;
    int tileK = tile % TK;
    int tileN = tile / TK;
    int hi = lane >> 4;
    int n  = lane & 15;

    int k0  = tileK * 32 + kmap_half(j2, hi);   // j2,j2+1 -> k0,k0+1
    int col = tileN * 16 + n;

    const float* s = src + (size_t)e * K * N;
    float w0 = s[(size_t)k0 * N + col];
    float w1 = s[(size_t)(k0 + 1) * N + col];
    if (norm_w) {                               // fold RMSNorm weight (K==kD)
        w0 *= norm_w[e * kD + k0];
        w1 *= norm_w[e * kD + k0 + 1];
    }
    dst[idx] = ((unsigned int)bf16_bits(w1) << 16) | bf16_bits(w0);
}

// ---------------------------------------------------------------------------
// Kernel 3: routed SwiGLU FFN. Block = 8 wave32, M = 64 token-pairs
// (4 M-subtiles per wave share every B fragment -> B traffic / 4).
// GEMM1 -> SiLU*gate -> 64x2048 bf16 h tile entirely in LDS (256KB, dynamic)
// -> GEMM2 -> gate-scaled f32 atomic combine into out.
// SWZ=true : B fragments are two contiguous global_load_b128 of preswizzled
//            bf16 weights.  SWZ=false: gather fp32 + convert on the fly.
// ---------------------------------------------------------------------------
template <bool SWZ>
__global__ __launch_bounds__(256) void moe_ffn_kernel(
    const __bf16* __restrict__ xh,
    const float* __restrict__ norm_w,
    const float* __restrict__ W1, const float* __restrict__ b1,
    const float* __restrict__ W2, const float* __restrict__ b2,
    const float* __restrict__ W3, const float* __restrict__ b3,
    const __bf16* __restrict__ W1s, const __bf16* __restrict__ W2s,
    const __bf16* __restrict__ W3s,
    const float* __restrict__ pairGate,
    const int* __restrict__ cnt, const int* __restrict__ lists,
    float* __restrict__ out)
{
    extern __shared__ char smem[];
    __bf16* sH    = (__bf16*)smem;                       // 64 x kF (256 KB)
    int*    sTok  = (int*)(smem + (size_t)64 * kF * 2);
    float*  sGate = (float*)(sTok + 64);
    int*    sValid= (int*)(sGate + 64);

    int e    = blockIdx.x >> 6;          // expert
    int tile = blockIdx.x & 63;          // 64-row tile in expert's list
    int n_e  = cnt[e];
    int row0 = tile * 64;
    if (row0 >= n_e) return;             // uniform: whole block exits

    int tid = threadIdx.x;
    if (tid < 64) {
        int r = row0 + tid;
        if (r < n_e) {
            int p = lists[e * kT + r];
            sTok[tid]   = p >> 1;
            sGate[tid]  = pairGate[p];
            sValid[tid] = 1;
        } else {
            sTok[tid] = 0; sGate[tid] = 0.f; sValid[tid] = 0;
        }
    }
    __syncthreads();

    int wave = tid >> 5, lane = tid & 31;
    int hi = lane >> 4, ln = lane & 15;

    const float* w1e = W1 + (size_t)e * kD * kF;
    const float* w2e = W2 + (size_t)e * kD * kF;
    const float* w3e = W3 + (size_t)e * kF * kD;
    const float* nwe = norm_w + (size_t)e * kD;

    // A-row pointers for the 4 M-subtiles this lane reads (row = s*16 + ln)
    const unsigned int* xrow[4];
    #pragma unroll
    for (int s = 0; s < 4; s++)
        xrow[s] = (const unsigned int*)(xh + (size_t)sTok[s * 16 + ln] * kD);

    // ---- GEMM1: [64 x D] @ [D x F] (W1', W2'); 16 passes of 128 cols ----
    for (int pass = 0; pass < 16; pass++) {
        int col   = pass * 128 + wave * 16 + ln;
        int tileN = pass * 8 + wave;
        v8f accA[4], accG[4];
        #pragma unroll
        for (int s = 0; s < 4; s++) {
            accA[s] = (v8f){0.f,0.f,0.f,0.f,0.f,0.f,0.f,0.f};
            accG[s] = (v8f){0.f,0.f,0.f,0.f,0.f,0.f,0.f,0.f};
        }
        // swizzled tile stream: contiguous 1KB tiles along K
        const uint4* b1p = (const uint4*)(W1s + ((size_t)e * kD * kF)
                              + (size_t)tileN * (kD / 32) * 512 + lane * 16);
        const uint4* b2p = (const uint4*)(W2s + ((size_t)e * kD * kF)
                              + (size_t)tileN * (kD / 32) * 512 + lane * 16);

        for (int kk = 0; kk < kD; kk += 32) {
            Frag bA, bB;
            if (SWZ) {
                uint4 q0 = b1p[0], q1 = b1p[1];
                bA.u[0]=q0.x; bA.u[1]=q0.y; bA.u[2]=q0.z; bA.u[3]=q0.w;
                bA.u[4]=q1.x; bA.u[5]=q1.y; bA.u[6]=q1.z; bA.u[7]=q1.w;
                uint4 p0 = b2p[0], p1 = b2p[1];
                bB.u[0]=p0.x; bB.u[1]=p0.y; bB.u[2]=p0.z; bB.u[3]=p0.w;
                bB.u[4]=p1.x; bB.u[5]=p1.y; bB.u[6]=p1.z; bB.u[7]=p1.w;
                b1p += 64;  // next 1KB tile (32 K)
                b2p += 64;
            } else {
                #pragma unroll
                for (int j2 = 0; j2 < 16; j2++) {
                    int k = kk + kmap_half(j2, hi);
                    float nw = nwe[k];
                    bA.h[j2] = (__bf16)(w1e[(size_t)k * kF + col] * nw);
                    bB.h[j2] = (__bf16)(w2e[(size_t)k * kF + col] * nw);
                }
            }
            #pragma unroll
            for (int s = 0; s < 4; s++) {
                Frag a;
                #pragma unroll
                for (int j = 0; j < 8; j++)
                    a.u[j] = xrow[s][(kk + kmap_dword(j, hi)) >> 1];
                accA[s] = __builtin_amdgcn_wmma_f32_16x16x32_bf16(
                    false, a.v, false, bA.v, (short)0, accA[s], false, false);
                accG[s] = __builtin_amdgcn_wmma_f32_16x16x32_bf16(
                    false, a.v, false, bB.v, (short)0, accG[s], false, false);
            }
        }

        float bb1 = b1[e * kF + col];
        float bb2 = b2[e * kF + col];
        #pragma unroll
        for (int s = 0; s < 4; s++) {
            #pragma unroll
            for (int r = 0; r < 8; r++) {
                float aa = accA[s][r] + bb1;
                float gg = accG[s][r] + bb2;
                // silu(a)*g with fast v_rcp_f32 (no IEEE div chain)
                float hh = aa * __builtin_amdgcn_rcpf(1.f + __expf(-aa)) * gg;
                int m = s * 16 + r + hi * 8;              // C layout row
                sH[(size_t)m * kF + col] = (__bf16)hh;
            }
        }
    }
    __syncthreads();

    // ---- GEMM2: [64 x F] @ [F x D]; 8 passes of 128 cols ----
    const unsigned int* hrow[4];
    #pragma unroll
    for (int s = 0; s < 4; s++)
        hrow[s] = (const unsigned int*)(sH + (size_t)(s * 16 + ln) * kF);

    for (int pass = 0; pass < 8; pass++) {
        int col   = pass * 128 + wave * 16 + ln;
        int tileN = pass * 8 + wave;
        v8f acc[4];
        #pragma unroll
        for (int s = 0; s < 4; s++)
            acc[s] = (v8f){0.f,0.f,0.f,0.f,0.f,0.f,0.f,0.f};

        const uint4* b3p = (const uint4*)(W3s + ((size_t)e * kF * kD)
                              + (size_t)tileN * (kF / 32) * 512 + lane * 16);

        for (int kk = 0; kk < kF; kk += 32) {
            Frag b;
            if (SWZ) {
                uint4 q0 = b3p[0], q1 = b3p[1];
                b.u[0]=q0.x; b.u[1]=q0.y; b.u[2]=q0.z; b.u[3]=q0.w;
                b.u[4]=q1.x; b.u[5]=q1.y; b.u[6]=q1.z; b.u[7]=q1.w;
                b3p += 64;
            } else {
                #pragma unroll
                for (int j2 = 0; j2 < 16; j2++) {
                    int k = kk + kmap_half(j2, hi);
                    b.h[j2] = (__bf16)w3e[(size_t)k * kD + col];
                }
            }
            #pragma unroll
            for (int s = 0; s < 4; s++) {
                Frag a;
                #pragma unroll
                for (int j = 0; j < 8; j++)
                    a.u[j] = hrow[s][(kk + kmap_dword(j, hi)) >> 1]; // ds reads
                acc[s] = __builtin_amdgcn_wmma_f32_16x16x32_bf16(
                    false, a.v, false, b.v, (short)0, acc[s], false, false);
            }
        }

        float bb3 = b3[e * kD + col];
        #pragma unroll
        for (int s = 0; s < 4; s++) {
            #pragma unroll
            for (int r = 0; r < 8; r++) {
                int m = s * 16 + r + hi * 8;
                if (sValid[m]) {
                    float val = sGate[m] * (acc[s][r] + bb3);
                    // two commutative f32 adds per element -> deterministic
                    unsafeAtomicAdd(&out[(size_t)sTok[m] * kD + col], val);
                }
            }
        }
    }
}

// ---------------------------------------------------------------------------
extern "C" void kernel_launch(void* const* d_in, const int* in_sizes, int n_in,
                              void* d_out, int out_size, void* d_ws, size_t ws_size,
                              hipStream_t stream)
{
    const float* x      = (const float*)d_in[0];
    const float* Wg     = (const float*)d_in[1];
    const float* norm_w = (const float*)d_in[2];
    const float* W1     = (const float*)d_in[3];
    const float* b1     = (const float*)d_in[4];
    const float* W2     = (const float*)d_in[5];
    const float* b2     = (const float*)d_in[6];
    const float* W3     = (const float*)d_in[7];
    const float* b3     = (const float*)d_in[8];
    float* out = (float*)d_out;

    // workspace layout (bytes)
    char* ws = (char*)d_ws;
    const size_t MB = 1024 * 1024;
    __bf16* xh       = (__bf16*)(ws + 0);                 // T*D*2 = 8 MB
    float*  pairGate = (float*)(ws + 8 * MB);             // 32 KB
    int*    pairExp  = (int*)(ws + 8 * MB + 32768);       // 32 KB
    int*    cnt      = (int*)(ws + 8 * MB + 65536);       // 256 B
    int*    lists    = (int*)(ws + 8 * MB + 65536 + 256); // E*T*4 = 128 KB
    __bf16* W1s      = (__bf16*)(ws + 16 * MB);           // 32 MB each
    __bf16* W2s      = (__bf16*)(ws + 48 * MB);
    __bf16* W3s      = (__bf16*)(ws + 80 * MB);
    const size_t WS_NEEDED = 112 * MB;
    const bool swz = (ws_size >= WS_NEEDED);

    hipMemsetAsync(d_out, 0, (size_t)out_size * sizeof(float), stream);
    hipMemsetAsync(cnt, 0, 256, stream);

    gate_norm_kernel<<<kT / 8, 256, 0, stream>>>(x, Wg, xh, pairGate, pairExp);
    build_lists_kernel<<<(kT + 255) / 256, 256, 0, stream>>>(pairExp, cnt, lists);

    const size_t SMEM = (size_t)64 * kF * 2 + 64 * 12;    // 256 KB h + row meta

    if (swz) {
        int blk = (int)(((size_t)kE * kD * kF / 2 + 255) / 256);
        convert_weights_kernel<<<blk, 256, 0, stream>>>(
            W1, norm_w, (unsigned int*)W1s, kD, kF);
        convert_weights_kernel<<<blk, 256, 0, stream>>>(
            W2, norm_w, (unsigned int*)W2s, kD, kF);
        convert_weights_kernel<<<blk, 256, 0, stream>>>(
            W3, nullptr, (unsigned int*)W3s, kF, kD);
        moe_ffn_kernel<true><<<kE * 64, 256, SMEM, stream>>>(
            xh, norm_w, W1, b1, W2, b2, W3, b3, W1s, W2s, W3s,
            pairGate, cnt, lists, out);
    } else {
        moe_ffn_kernel<false><<<kE * 64, 256, SMEM, stream>>>(
            xh, norm_w, W1, b1, W2, b2, W3, b3, W1s, W2s, W3s,
            pairGate, cnt, lists, out);
    }
}
